// BahdanauAttention_75187697484051
// MI455X (gfx1250) — compile-verified, matrix operand order checked by reading, not run on hardware
//
#include <hip/hip_runtime.h>
#include <hip/hip_bf16.h>

// B=32, S=4096, D=256, UNITS=256
#define BB   32
#define SS   4096
#define DD   256
#define UU   256

typedef __attribute__((ext_vector_type(16))) __bf16          v16bf;
typedef __attribute__((ext_vector_type(16))) unsigned short  v16u;
typedef __attribute__((ext_vector_type(8)))  float           v8f;

// Native bf16 convert on gfx1250 (RNE) -> v_cvt_pk_bf16_f32.
static __device__ __forceinline__ __bf16 to_bf16(float f) {
    return (__bf16)f;
}

// Native single-instruction tanh on gfx1250 when available.
#if __has_builtin(__builtin_amdgcn_tanhf)
#define FAST_TANH(x) __builtin_amdgcn_tanhf(x)
#elif __has_builtin(__builtin_amdgcn_tanh_f32)
#define FAST_TANH(x) __builtin_amdgcn_tanh_f32(x)
#else
#define FAST_TANH(x) tanhf(x)
#endif

// ---------------------------------------------------------------------------
// Kernel 1: q_proj = query @ W2  (fp32)  +  swizzle W1 -> bf16 B-fragments
// B-fragment layout (ISA 7.12.2, 16-bit 32x16 B): lane n (0-15) holds col n,
// K = {kbase..kbase+7, kbase+16..kbase+23}, kbase = kt*32 (+8 for lanes 16-31).
// Packed: w1swz[((kt*16+nt)*32+lane)*16 + j]   (16 bf16 = 32B contiguous/lane)
// ---------------------------------------------------------------------------
__global__ void bah_prep_kernel(const float* __restrict__ query,
                                const float* __restrict__ W2,
                                const float* __restrict__ W1,
                                float* __restrict__ qproj,
                                unsigned short* __restrict__ w1swz) {
    int idx = blockIdx.x * blockDim.x + threadIdx.x;
    if (idx < BB * UU) {
        int b = idx >> 8, u = idx & 255;
        float acc = 0.f;
        for (int d = 0; d < DD; ++d)
            acc += query[b * DD + d] * W2[d * UU + u];
        qproj[idx] = acc;
    } else if (idx < BB * UU + 8 * 16 * 32) {
        int t    = idx - BB * UU;
        int lane = t & 31;
        int nt   = (t >> 5) & 15;
        int kt   = t >> 9;
        int col   = nt * 16 + (lane & 15);
        int kbase = kt * 32 + ((lane & 16) ? 8 : 0);
        v16bf pk;
#pragma unroll
        for (int j = 0; j < 16; ++j) {
            int k = kbase + (j < 8 ? j : 8 + j);   // j>=8 -> kbase+16+(j-8)
            pk[j] = to_bf16(W1[k * UU + col]);
        }
        *(v16u*)(w1swz + (size_t)t * 16) = __builtin_bit_cast(v16u, pk);
    }
}

// ---------------------------------------------------------------------------
// Kernel 2: scores[b, s] = tanh(values[b,s,:] @ W1 + qproj[b,:]) . V
// One wave per (batch, 32-row tile pair of S): each B fragment loaded once
// feeds TWO WMMAs (two 16-row A tiles), halving W1-fragment L2 traffic.
// bf16 WMMA 16x16x32, f32 accumulate.
// ---------------------------------------------------------------------------
__global__ void __launch_bounds__(256)
bah_score_kernel(const float* __restrict__ values,
                 const unsigned short* __restrict__ w1swz,
                 const float* __restrict__ qproj,
                 const float* __restrict__ V,
                 float* __restrict__ scores) {
    const int lane = threadIdx.x & 31;
    const int wave = (blockIdx.x * blockDim.x + threadIdx.x) >> 5;
    const int b    = wave >> 7;           // 128 tile-pairs per batch
    const int s0   = (wave & 127) << 5;   // 32 rows per wave

    // --- load A fragments: two 16x256 tiles -> 2 x 8 k-tiles of 16x32 bf16 ---
    // A layout (16-bit 16x32): lane l(0-15) = row l, K {0..7,16..23};
    // lane l+16 = row l, K {8..15,24..31}.
    const float* vrow0 = values + ((size_t)(b * SS + s0 +      (lane & 15))) * DD;
    const float* vrow1 = values + ((size_t)(b * SS + s0 + 16 + (lane & 15))) * DD;
    const int koff = (lane & 16) ? 8 : 0;
    v16bf a0[8], a1[8];
#pragma unroll
    for (int kt = 0; kt < 8; ++kt) {
        const int kbase = kt * 32 + koff;
        v16bf p0, p1;
#pragma unroll
        for (int j = 0; j < 8; ++j) {
            p0[j]     = to_bf16(vrow0[kbase + j]);
            p0[8 + j] = to_bf16(vrow0[kbase + 16 + j]);
            p1[j]     = to_bf16(vrow1[kbase + j]);
            p1[8 + j] = to_bf16(vrow1[kbase + 16 + j]);
        }
        a0[kt] = p0;
        a1[kt] = p1;
    }

    // Per-lane partials: each lane accumulates its own column's tanh(.)*V
    // contribution across all 16 unit tiles; butterfly-reduce once at the end.
    float rs0[8], rs1[8];
#pragma unroll
    for (int r = 0; r < 8; ++r) { rs0[r] = 0.f; rs1[r] = 0.f; }

    const v16u* bfrag = (const v16u*)w1swz;
#pragma unroll 1
    for (int nt = 0; nt < 16; ++nt) {
        v8f c0 = {}, c1 = {};
#pragma unroll
        for (int kt = 0; kt < 8; ++kt) {
            v16bf bb = __builtin_bit_cast(v16bf, bfrag[(kt * 16 + nt) * 32 + lane]);
            c0 = __builtin_amdgcn_wmma_f32_16x16x32_bf16(
                     false, a0[kt], false, bb, (short)0, c0, false, false);
            c1 = __builtin_amdgcn_wmma_f32_16x16x32_bf16(
                     false, a1[kt], false, bb, (short)0, c1, false, false);
        }
        // C layout: VGPR r, lane<16 -> (M=r, N=lane); lane>=16 -> (M=r+8, N=lane-16)
        const int   u  = nt * 16 + (lane & 15);
        const float q  = qproj[b * UU + u];
        const float vv = V[u];
#pragma unroll
        for (int r = 0; r < 8; ++r) {
            rs0[r] += FAST_TANH(c0[r] + q) * vv;
            rs1[r] += FAST_TANH(c1[r] + q) * vv;
        }
    }

    // Sum over the 16 columns (butterfly stays inside each 16-lane half).
#pragma unroll
    for (int r = 0; r < 8; ++r) {
        float t0 = rs0[r], t1 = rs1[r];
        t0 += __shfl_xor(t0, 1);  t1 += __shfl_xor(t1, 1);
        t0 += __shfl_xor(t0, 2);  t1 += __shfl_xor(t1, 2);
        t0 += __shfl_xor(t0, 4);  t1 += __shfl_xor(t1, 4);
        t0 += __shfl_xor(t0, 8);  t1 += __shfl_xor(t1, 8);
        rs0[r] = t0;  rs1[r] = t1;
    }

    // lane 0 holds rows {s0+0..7, s0+16..23}, lane 16 holds rows {+8..15, +24..31}
    if ((lane & 15) == 0) {
        const int mbase = (lane >> 4) << 3;
#pragma unroll
        for (int r = 0; r < 8; ++r) {
            scores[b * SS + s0 + mbase + r]      = rs0[r];
            scores[b * SS + s0 + 16 + mbase + r] = rs1[r];
        }
    }
}

// ---------------------------------------------------------------------------
// Kernel 3: in-place softmax over S per batch (weights live in d_out)
// ---------------------------------------------------------------------------
__global__ void bah_softmax_kernel(float* __restrict__ w) {
    const int b = blockIdx.x;
    const int t = threadIdx.x;            // 256 threads
    __shared__ float red[256];
    float* s = w + (size_t)b * SS;

    float mx = -3.402823466e+38f;
    for (int i = t; i < SS; i += 256) mx = fmaxf(mx, s[i]);
    red[t] = mx; __syncthreads();
    for (int st = 128; st > 0; st >>= 1) {
        if (t < st) red[t] = fmaxf(red[t], red[t + st]);
        __syncthreads();
    }
    mx = red[0]; __syncthreads();

    float sum = 0.f;
    for (int i = t; i < SS; i += 256) sum += __expf(s[i] - mx);
    red[t] = sum; __syncthreads();
    for (int st = 128; st > 0; st >>= 1) {
        if (t < st) red[t] += red[t + st];
        __syncthreads();
    }
    const float inv = 1.0f / red[0];

    for (int i = t; i < SS; i += 256) s[i] = __expf(s[i] - mx) * inv;
}

// ---------------------------------------------------------------------------
// Kernel 4: zero the context region (atomics accumulate per call)
// ---------------------------------------------------------------------------
__global__ void bah_zero_kernel(float* __restrict__ ctx) {
    int i = blockIdx.x * blockDim.x + threadIdx.x;
    if (i < BB * DD) ctx[i] = 0.f;
}

// ---------------------------------------------------------------------------
// Kernel 5: context[b,d] = sum_s w[b,s] * values[b,s,d]
// Block = (batch, S-chunk of 512); coalesced over d; f32 global atomic add.
// Second pass over values hits the 192 MB L2 (values = 134 MB).
// ---------------------------------------------------------------------------
__global__ void bah_context_kernel(const float* __restrict__ values,
                                   const float* __restrict__ weights,
                                   float* __restrict__ ctx) {
    const int blk   = blockIdx.x;
    const int b     = blk >> 3;
    const int chunk = blk & 7;
    const int d     = threadIdx.x;        // 256 threads

    const float* w = weights + (size_t)b * SS + chunk * 512;
    const float* v = values + ((size_t)b * SS + (size_t)chunk * 512) * DD + d;

    float acc = 0.f;
    for (int s = 0; s < 512; ++s)
        acc = fmaf(w[s], v[(size_t)s * DD], acc);

    atomicAdd(&ctx[b * DD + d], acc);
}

// ---------------------------------------------------------------------------
extern "C" void kernel_launch(void* const* d_in, const int* in_sizes, int n_in,
                              void* d_out, int out_size, void* d_ws, size_t ws_size,
                              hipStream_t stream) {
    const float* query  = (const float*)d_in[0];   // [32,256]
    const float* values = (const float*)d_in[1];   // [32,4096,256]
    const float* W1     = (const float*)d_in[2];   // [256,256]
    const float* W2     = (const float*)d_in[3];   // [256,256]
    const float* V      = (const float*)d_in[4];   // [256,1]

    float* ctx     = (float*)d_out;                // [32,256]  (output 0)
    float* weights = (float*)d_out + BB * DD;      // [32,4096] (output 1)

    // workspace: qproj (32 KB) + swizzled bf16 W1 fragments (128 KB)
    float*          qproj = (float*)d_ws;
    unsigned short* w1swz = (unsigned short*)((char*)d_ws + BB * UU * sizeof(float));

    // 1) q_proj + W1 swizzle (8192 + 4096 threads)
    bah_prep_kernel<<<48, 256, 0, stream>>>(query, W2, W1, qproj, w1swz);

    // 2) scores via bf16 WMMA: 4096 waves (32 rows each) = 512 blocks x 256
    bah_score_kernel<<<512, 256, 0, stream>>>(values, w1swz, qproj, V, weights);

    // 3) softmax in place over S, per batch
    bah_softmax_kernel<<<BB, 256, 0, stream>>>(weights);

    // 4) zero context accumulators
    bah_zero_kernel<<<(BB * DD + 255) / 256, 256, 0, stream>>>(ctx);

    // 5) context = weights . values
    bah_context_kernel<<<BB * 8, 256, 0, stream>>>(values, weights, ctx);
}
